// SparseMHADecoder_3582002725463
// MI455X (gfx1250) — matrix operand order repaired and checked by reading, not compiled, not run on hardware
//
#include <hip/hip_runtime.h>
#include <hip/hip_bf16.h>

// ---------------------------------------------------------------------------
// Problem constants (fixed by the reference)
// ---------------------------------------------------------------------------
#define B_       2
#define LQ_      4096
#define LKV_     2048
#define H_       16
#define DH_      64       // DQK == DHV == 64
#define DMODEL_  1024     // H_*DH_ == DQ == DK == DV == DOUT
#define MAXSPAN_ 16       // P row stride (span <= 16 in this config)

// ---------------------------------------------------------------------------
// WMMA types (gfx1250, wave32)
// ---------------------------------------------------------------------------
typedef __bf16 bf16_t;
typedef bf16_t v16bf __attribute__((ext_vector_type(16)));
typedef float  v8f   __attribute__((ext_vector_type(8)));

__device__ __forceinline__ unsigned short f2bf(float f) {
    union { float f; unsigned u; } x; x.f = f;
    unsigned r = x.u + 0x7FFFu + ((x.u >> 16) & 1u);   // round-to-nearest-even
    return (unsigned short)(r >> 16);
}

// ---------------------------------------------------------------------------
// Pre-pass 1: elementwise fp32 -> bf16 (activations)
// ---------------------------------------------------------------------------
__global__ __launch_bounds__(256)
void conv_f32_bf16_kernel(const float* __restrict__ src,
                          unsigned short* __restrict__ dst) {
    size_t i = (size_t)blockIdx.x * blockDim.x + threadIdx.x;   // per float4
    float4 f = *(const float4*)(src + i * 4);
    ushort4 o;
    o.x = f2bf(f.x); o.y = f2bf(f.y); o.z = f2bf(f.z); o.w = f2bf(f.w);
    *(ushort4*)(dst + i * 4) = o;
}

// ---------------------------------------------------------------------------
// Pre-pass 2: W[K,N] fp32 -> Wt[N,K] bf16 (transpose-convert, LDS tiled)
// Grid (N/32, K/32), block (32, 8).
// ---------------------------------------------------------------------------
__global__ __launch_bounds__(256)
void transpose_conv_kernel(const float* __restrict__ W,
                           unsigned short* __restrict__ Wt,
                           int K, int N) {
    __shared__ unsigned short tile[32][33];
    const int x  = threadIdx.x;          // 0..31
    const int y  = threadIdx.y;          // 0..7
    const int n0 = blockIdx.x * 32;
    const int k0 = blockIdx.y * 32;
#pragma unroll
    for (int r = 0; r < 4; ++r)
        tile[y + 8 * r][x] = f2bf(W[(size_t)(k0 + y + 8 * r) * N + (n0 + x)]);
    __syncthreads();
#pragma unroll
    for (int r = 0; r < 4; ++r)
        Wt[(size_t)(n0 + y + 8 * r) * K + (k0 + x)] = tile[x][y + 8 * r];
}

// ---------------------------------------------------------------------------
// Dense GEMM: C[M,N] (fp32) = A[M,K] (bf16, row-major) @ Bt[N,K]^T (bf16).
// Block tile 128x128, K-step 32, 256 threads = 8 waves (4 Mwaves x 2 Nwaves),
// each wave: 2x4 v_wmma_f32_16x16x32_bf16 accumulators.
//
// Data movement: GLOBAL_LOAD_ASYNC_TO_LDS_B128 (ASYNCcnt), triple-buffered
// LDS, fragments land directly in WMMA register layout (ISA 7.12.2):
//   A lane l of subtile s: row m = s*16+(l&15); elems 0-7 = K (l>>4)*8+0..7,
//     elems 8-15 = K 16+(l>>4)*8+0..7 -> two contiguous 16B chunks of A.
//   B lane l of subtile s: col n = s*16+(l&15); elems 0-7 = K (l>>4)*8+0..7,
//     elems 8-15 = K 16+(l>>4)*8+0..7 -> two contiguous 16B chunks of Bt row n.
// Per K-step per wave: 4 async instructions (32 lanes x 16B each) = 2KB.
// ---------------------------------------------------------------------------
#define BM 128
#define BN 128
#define BK 32
#define TILE_BYTES 16384            // 8KB A + 8KB B per buffer
#define NBUF 3

__device__ __forceinline__ void async_b128(unsigned lds_byte, unsigned g_byte,
                                           unsigned long long base) {
    asm volatile("global_load_async_to_lds_b128 %0, %1, %2"
                 :: "v"(lds_byte), "v"(g_byte), "s"(base)
                 : "memory");
}

__global__ __launch_bounds__(256)
void wmma_gemm_bf16_kernel(const unsigned short* __restrict__ A,
                           const unsigned short* __restrict__ Bt,
                           float* __restrict__ C,
                           int M, int N, int K) {
    extern __shared__ __align__(16) char smem[];   // NBUF * TILE_BYTES

    const int tid   = threadIdx.x;
    const int lane  = tid & 31;
    const int wave  = tid >> 5;
    const int waveM = wave >> 1;   // 0..3
    const int waveN = wave & 1;    // 0..1

    const unsigned blockN0 = blockIdx.x * BN;
    const unsigned blockM0 = blockIdx.y * BM;
    const unsigned long long Abase = (unsigned long long)A;
    const unsigned long long Bbase = (unsigned long long)Bt;
    const unsigned Ku = (unsigned)K;

    v8f acc[2][4];
#pragma unroll
    for (int i = 0; i < 2; ++i)
#pragma unroll
        for (int j = 0; j < 4; ++j)
            acc[i][j] = (v8f){0.f,0.f,0.f,0.f,0.f,0.f,0.f,0.f};

    // Issue one 16KB tile (A 8KB + B 8KB) with 4 async instructions per wave.
    auto issue_tile = [&](unsigned kk, int buf) {
        unsigned bufbase = (unsigned)buf * TILE_BYTES;
        // A chunks: chunk id ch in [0,512): s=ch>>6, l=(ch>>1)&31, c=ch&1
        unsigned ch = (unsigned)tid;
#pragma unroll
        for (int rep = 0; rep < 2; ++rep, ch += 256) {
            unsigned l  = (ch >> 1) & 31u;
            unsigned m  = blockM0 + ((ch >> 6) << 4) + (l & 15u);
            unsigned kl = ((ch & 1u) << 4) + ((l >> 4) << 3);
            async_b128(bufbase + ch * 16u, (m * Ku + kk + kl) * 2u, Abase);
        }
        // B chunks
        ch = (unsigned)tid;
#pragma unroll
        for (int rep = 0; rep < 2; ++rep, ch += 256) {
            unsigned l  = (ch >> 1) & 31u;
            unsigned n  = blockN0 + ((ch >> 6) << 4) + (l & 15u);
            unsigned kl = ((ch & 1u) << 4) + ((l >> 4) << 3);
            async_b128(bufbase + 8192u + ch * 16u, (n * Ku + kk + kl) * 2u, Bbase);
        }
    };

    const int nk = K / BK;
    int cur = 0, nxt = 1;

    issue_tile(0u, 0);                          // prologue: tile 0 -> buf 0

    for (int i = 0; i < nk; ++i) {
        if (i + 1 < nk) {
            issue_tile((unsigned)(i + 1) * BK, nxt);
            asm volatile("s_wait_asynccnt 0x4" ::: "memory");  // tile i landed
        } else {
            asm volatile("s_wait_asynccnt 0x0" ::: "memory");
        }
        __syncthreads();   // all waves' tile-i data visible in LDS

        const char* bufA = smem + cur * TILE_BYTES;
        const char* bufB = bufA + 8192;

        v16bf afrag[2], bfrag[4];
#pragma unroll
        for (int a = 0; a < 2; ++a)
            afrag[a] = *(const v16bf*)(bufA + (waveM * 2 + a) * 1024 + lane * 32);
#pragma unroll
        for (int b = 0; b < 4; ++b)
            bfrag[b] = *(const v16bf*)(bufB + (waveN * 4 + b) * 1024 + lane * 32);

#pragma unroll
        for (int a = 0; a < 2; ++a)
#pragma unroll
            for (int b = 0; b < 4; ++b)
                acc[a][b] = __builtin_amdgcn_wmma_f32_16x16x32_bf16(
                    false, afrag[a], false, bfrag[b],
                    (short)0, acc[a][b], false, false);

        cur = nxt;
        nxt = (nxt == NBUF - 1) ? 0 : nxt + 1;
    }

    // Writeback: C/D layout = VGPR r, lanes 0-15 -> M=r, lanes 16-31 -> M=8+r
#pragma unroll
    for (int a = 0; a < 2; ++a) {
#pragma unroll
        for (int b = 0; b < 4; ++b) {
            int row0 = blockM0 + (waveM * 2 + a) * 16 + ((lane >> 4) << 3);
            int col  = blockN0 + (waveN * 4 + b) * 16 + (lane & 15);
            float* crow = C + (size_t)row0 * N + col;
#pragma unroll
            for (int r = 0; r < 8; ++r)
                crow[(size_t)r * N] = acc[a][b][r];
        }
    }
}

// ---------------------------------------------------------------------------
// Pass 1: banded scores + softmax over the QUERY axis (axis=2, faithful).
// Each KV column c sees rows q = c*stride + r, r in [0, span).
// One block = 16 columns x 16 row-slots; P[b,h,c,r] = normalized probability.
// ---------------------------------------------------------------------------
__global__ __launch_bounds__(256)
void attn_colsoftmax_kernel(const float* __restrict__ Q,
                            const float* __restrict__ Kp,
                            float* __restrict__ P,
                            const int* __restrict__ spanp,
                            const int* __restrict__ stridep) {
    const int span   = *spanp;
    const int stride = *stridep;

    __shared__ float sc[16][17];

    const int tid = threadIdx.x;
    const int r   = tid & 15;
    const int cl  = tid >> 4;

    const int cid = blockIdx.x * 16 + cl;           // over B*H*LKV
    const int c   = cid & (LKV_ - 1);
    const int h   = (cid / LKV_) & (H_ - 1);
    const int b   = cid / (LKV_ * H_);

    const int qrow = c * stride + r;

    float s = -__builtin_inff();
    if (r < span && qrow < LQ_) {
        const float* qv = Q  + ((size_t)(b * LQ_  + qrow) * H_ + h) * DH_;
        const float* kv = Kp + ((size_t)(b * LKV_ + c)    * H_ + h) * DH_;
        float acc = 0.f;
#pragma unroll
        for (int d = 0; d < DH_; d += 4) {
            float4 a  = *(const float4*)(qv + d);
            float4 kk = *(const float4*)(kv + d);
            acc = fmaf(a.x, kk.x, acc);
            acc = fmaf(a.y, kk.y, acc);
            acc = fmaf(a.z, kk.z, acc);
            acc = fmaf(a.w, kk.w, acc);
        }
        s = acc;
    }
    sc[cl][r] = s;
    __syncthreads();

    float m = -__builtin_inff();
#pragma unroll
    for (int i = 0; i < 16; ++i) m = fmaxf(m, sc[cl][i]);
    float l = 0.f;
#pragma unroll
    for (int i = 0; i < 16; ++i) {
        float si = sc[cl][i];
        l += (si == -__builtin_inff()) ? 0.f : __expf(si - m);
    }

    float p = 0.f;
    if (s != -__builtin_inff()) p = __expf(s - m) / l;
    P[(size_t)cid * MAXSPAN_ + r] = p;
}

// ---------------------------------------------------------------------------
// Pass 2: ctx[b,q,h,d] = sum_c P[b,h,c,q-c*stride] * V[b,c,h,d], bf16 output
// (feeds the final async-WMMA GEMM directly). One thread per output element.
// ---------------------------------------------------------------------------
__global__ __launch_bounds__(256)
void attn_ctx_kernel(const float* __restrict__ P,
                     const float* __restrict__ Vp,
                     unsigned short* __restrict__ ctx,
                     const int* __restrict__ spanp,
                     const int* __restrict__ stridep) {
    const int span   = *spanp;
    const int stride = *stridep;

    const int gid  = blockIdx.x * blockDim.x + threadIdx.x;
    const int d    = gid & (DH_ - 1);
    const int h    = (gid >> 6) & (H_ - 1);
    const int qrow = (gid >> 10) & (LQ_ - 1);
    const int b    = gid >> 22;

    int lo   = qrow - span + 1;
    int cmin = (lo > 0) ? (lo + stride - 1) / stride : 0;
    int cmax = qrow / stride;
    if (cmax > LKV_ - 1) cmax = LKV_ - 1;

    const float* pb = P + (size_t)((b * H_ + h) * LKV_) * MAXSPAN_;

    float acc = 0.f;
    for (int c = cmin; c <= cmax; ++c) {
        int r = qrow - c * stride;
        if (r >= 0 && r < span && r < MAXSPAN_) {
            float p = pb[(size_t)c * MAXSPAN_ + r];
            float v = Vp[((size_t)(b * LKV_ + c) * H_ + h) * DH_ + d];
            acc = fmaf(p, v, acc);
        }
    }
    ctx[((size_t)(b * LQ_ + qrow) * H_ + h) * DH_ + d] = f2bf(acc);
}

// ---------------------------------------------------------------------------
// Launcher
// ---------------------------------------------------------------------------
extern "C" void kernel_launch(void* const* d_in, const int* in_sizes, int n_in,
                              void* d_out, int out_size, void* d_ws, size_t ws_size,
                              hipStream_t stream) {
    const float* q  = (const float*)d_in[0];
    const float* k  = (const float*)d_in[1];
    const float* v  = (const float*)d_in[2];
    const float* Wq = (const float*)d_in[3];
    const float* Wk = (const float*)d_in[4];
    const float* Wv = (const float*)d_in[5];
    const float* Wo = (const float*)d_in[6];
    const int* span   = (const int*)d_in[7];
    const int* stride = (const int*)d_in[8];
    float* out = (float*)d_out;

    const int MQ = B_ * LQ_;    // 8192
    const int MK = B_ * LKV_;   // 4096

    // ---- workspace layout (bytes), with aliasing ----
    char* w = (char*)d_ws;
    unsigned short* qbf   = (unsigned short*)w;                 // 16 MB
    unsigned short* ctxbf = qbf;                                // alias (q dead)
    w += (size_t)MQ * DMODEL_ * 2;
    unsigned short* kbf = (unsigned short*)w;                   // 8 MB
    float*          Pp  = (float*)w;                            // alias (k dead)
    w += (size_t)MK * DMODEL_ * 2;
    unsigned short* vbf = (unsigned short*)w;  w += (size_t)MK * DMODEL_ * 2;
    unsigned short* Wqt = (unsigned short*)w;  w += (size_t)DMODEL_ * DMODEL_ * 2;
    unsigned short* Wkt = (unsigned short*)w;  w += (size_t)DMODEL_ * DMODEL_ * 2;
    unsigned short* Wvt = (unsigned short*)w;  w += (size_t)DMODEL_ * DMODEL_ * 2;
    unsigned short* Wot = (unsigned short*)w;  w += (size_t)DMODEL_ * DMODEL_ * 2;
    float* Qp = (float*)w;  w += (size_t)MQ * DMODEL_ * 4;
    float* Kp = (float*)w;  w += (size_t)MK * DMODEL_ * 4;
    float* Vp = (float*)w;  w += (size_t)MK * DMODEL_ * 4;

    const size_t LDSB = NBUF * TILE_BYTES;   // 48 KB dynamic LDS
    const dim3 tgrid(DMODEL_ / 32, DMODEL_ / 32);
    const dim3 tblk(32, 8);

    // ---- fp32 -> bf16 conversions (activations) and W^T (weights) ----
    conv_f32_bf16_kernel<<<(size_t)MQ * DMODEL_ / 1024, 256, 0, stream>>>(q, qbf);
    conv_f32_bf16_kernel<<<(size_t)MK * DMODEL_ / 1024, 256, 0, stream>>>(k, kbf);
    conv_f32_bf16_kernel<<<(size_t)MK * DMODEL_ / 1024, 256, 0, stream>>>(v, vbf);
    transpose_conv_kernel<<<tgrid, tblk, 0, stream>>>(Wq, Wqt, DMODEL_, DMODEL_);
    transpose_conv_kernel<<<tgrid, tblk, 0, stream>>>(Wk, Wkt, DMODEL_, DMODEL_);
    transpose_conv_kernel<<<tgrid, tblk, 0, stream>>>(Wv, Wvt, DMODEL_, DMODEL_);
    transpose_conv_kernel<<<tgrid, tblk, 0, stream>>>(Wo, Wot, DMODEL_, DMODEL_);

    // ---- projections: async-pipelined bf16 WMMA GEMMs ----
    wmma_gemm_bf16_kernel<<<dim3(DMODEL_ / BN, MQ / BM), 256, LDSB, stream>>>(
        qbf, Wqt, Qp, MQ, DMODEL_, DMODEL_);
    wmma_gemm_bf16_kernel<<<dim3(DMODEL_ / BN, MK / BM), 256, LDSB, stream>>>(
        kbf, Wkt, Kp, MK, DMODEL_, DMODEL_);
    wmma_gemm_bf16_kernel<<<dim3(DMODEL_ / BN, MK / BM), 256, LDSB, stream>>>(
        vbf, Wvt, Vp, MK, DMODEL_, DMODEL_);

    // ---- banded scores + query-axis softmax -> probabilities P ----
    attn_colsoftmax_kernel<<<(B_ * H_ * LKV_) / 16, 256, 0, stream>>>(
        Qp, Kp, Pp, span, stride);

    // ---- sparse context accumulation -> bf16 ctx (aliased over qbf) ----
    attn_ctx_kernel<<<(B_ * LQ_ * H_ * DH_) / 256, 256, 0, stream>>>(
        Pp, Vp, ctxbf, span, stride);

    // ---- output projection: out = ctx @ Wo ----
    wmma_gemm_bf16_kernel<<<dim3(DMODEL_ / BN, MQ / BM), 256, LDSB, stream>>>(
        ctxbf, Wot, out, MQ, DMODEL_, DMODEL_);
}
